// SelfAttention_23553600651440
// MI455X (gfx1250) — matrix operand order repaired and checked by reading, not compiled, not run on hardware
//
#include <hip/hip_runtime.h>
#include <hip/hip_bf16.h>
#include <math.h>

// ---------------------------------------------------------------------------
// Self-attention forward on MI455X (gfx1250), bf16 WMMA path.
//   b=4, t=2048, k=128, h=8.  All GEMMs via v_wmma_f32_16x16x32_bf16.
//   - B operands fragment-major in LDS: every WMMA operand = 2x ds_load_b128
//   - straight-copy tiles via GLOBAL_LOAD_ASYNC_TO_LDS_B128 (ASYNCcnt)
//   - double-buffered fragment streaming so WMMAs don't stall on dscnt 0
// ---------------------------------------------------------------------------

#define B_   4
#define T_   2048
#define D_   128
#define H_   8
#define HD_  (H_ * D_)          // 1024
#define NTOK (B_ * T_)          // 8192

typedef __attribute__((ext_vector_type(16))) __bf16 v16bf;
typedef __attribute__((ext_vector_type(8)))  float  v8f;

union Frag {
    v16bf    v;
    unsigned u[8];
};

// --- CDNA5 async global->LDS DMA path (ASYNCcnt) --------------------------
#if defined(__has_builtin)
#if __has_builtin(__builtin_amdgcn_global_load_async_to_lds_b128) && \
    __has_builtin(__builtin_amdgcn_s_wait_asynccnt)
#define HAVE_ASYNC_LDS 1
#endif
#endif

#ifdef HAVE_ASYNC_LDS
typedef __attribute__((__vector_size__(4 * sizeof(int)))) int gccint4;
typedef __attribute__((address_space(1))) gccint4 as1_int4;
typedef __attribute__((address_space(3))) gccint4 as3_int4;
#define ASYNC_CPY16(ldsp, gp) \
    __builtin_amdgcn_global_load_async_to_lds_b128( \
        (as1_int4*)(gp), (as3_int4*)(ldsp), 0, 0)
#define ASYNC_WAIT() __builtin_amdgcn_s_wait_asynccnt(0)
#else
#define ASYNC_CPY16(ldsp, gp) (*(uint4*)(ldsp) = *(const uint4*)(gp))
#define ASYNC_WAIT()
#endif

static __device__ __forceinline__ unsigned short f2bf(float f) {
    union { float f; unsigned u; } x; x.f = f;
    unsigned r = x.u + 0x7FFFu + ((x.u >> 16) & 1u);   // round-to-nearest-even
    return (unsigned short)(r >> 16);
}

#define WMMA_BF16(a, b, c) \
    __builtin_amdgcn_wmma_f32_16x16x32_bf16(false, (a), false, (b), (short)0, (c), false, false)

// ---------------------------------------------------------------------------
// Kernel 0: f32 -> bf16 (grid-stride)
// ---------------------------------------------------------------------------
__global__ void cvt_bf16_kernel(const float* __restrict__ src,
                                unsigned short* __restrict__ dst, int n) {
    for (int i = blockIdx.x * blockDim.x + threadIdx.x; i < n;
         i += gridDim.x * blockDim.x)
        dst[i] = f2bf(src[i]);
}

// ---------------------------------------------------------------------------
// Kernel 1: QKV projection.  out[b,h,t,d] = scale * (x @ W) reshaped.
// ---------------------------------------------------------------------------
__global__ __launch_bounds__(256)
void qkv_proj_kernel(const unsigned short* __restrict__ xb,
                     const unsigned short* __restrict__ Wb,
                     unsigned short* __restrict__ dst,   // [B][H][T][D]
                     float scale) {
    __shared__ unsigned short As[128][72];    // A: 64-wide K chunk, padded rows
    __shared__ unsigned       BpT[128][36];   // B: fragment-major [col][kpair]

    const int tid  = threadIdx.x;
    const int lane = tid & 31;
    const int wave = tid >> 5;
    const int m    = lane & 15;
    const int hi   = lane >> 4;
    const int m0   = blockIdx.x * 128;
    const int n0   = blockIdx.y * 128;

    v8f acc[8];
#pragma unroll
    for (int i = 0; i < 8; ++i) acc[i] = (v8f){};

    auto loadB = [&](Frag& f, int kk, int nt) {
        const unsigned* brow = &BpT[nt * 16 + m][0];
        int kpb = (kk >> 1) + hi * 4;
        *(uint4*)&f.u[0] = *(const uint4*)&brow[kpb];
        *(uint4*)&f.u[4] = *(const uint4*)&brow[kpb + 8];
    };

    for (int kc = 0; kc < 128; kc += 64) {
        __syncthreads();
        for (int i = tid; i < 1024; i += 256) {
            int r = i >> 3, c4 = i & 7;
            ASYNC_CPY16(&As[r][c4 * 8], &xb[(size_t)(m0 + r) * 128 + kc + c4 * 8]);
        }
        for (int i = tid; i < 2048; i += 256) {
            int kp = i >> 6, np = i & 63;
            unsigned a = *(const unsigned*)&Wb[(size_t)(kc + 2 * kp) * HD_ + n0 + 2 * np];
            unsigned b = *(const unsigned*)&Wb[(size_t)(kc + 2 * kp + 1) * HD_ + n0 + 2 * np];
            BpT[2 * np][kp]     = (a & 0xFFFFu) | (b << 16);
            BpT[2 * np + 1][kp] = (a >> 16) | (b & 0xFFFF0000u);
        }
        ASYNC_WAIT();
        __syncthreads();

        // hoist both A fragments for this K chunk
        Frag af[2];
        {
            const unsigned* rowp = (const unsigned*)&As[wave * 16 + m][0];
#pragma unroll
            for (int k2 = 0; k2 < 2; ++k2) {
                int ab = k2 * 16 + hi * 4;
                *(uint4*)&af[k2].u[0] = *(const uint4*)&rowp[ab];
                *(uint4*)&af[k2].u[4] = *(const uint4*)&rowp[ab + 8];
            }
        }
        // double-buffered B fragment stream over (kk, nt)
        Frag bfr[2];
        loadB(bfr[0], 0, 0);
#pragma unroll
        for (int i = 0; i < 16; ++i) {
            int nt = i & 7;
            if (i < 15) {
                int j = i + 1;
                loadB(bfr[j & 1], (j >> 3) * 32, j & 7);
            }
            acc[nt] = WMMA_BF16(af[i >> 3].v, bfr[i & 1].v, acc[nt]);
        }
    }

    // epilogue: scale, bf16, scatter to [b,h,t,d]
#pragma unroll
    for (int nt = 0; nt < 8; ++nt) {
        int col = n0 + nt * 16 + m;
        int hh = col >> 7, dd = col & 127;
#pragma unroll
        for (int r = 0; r < 8; ++r) {
            int g  = m0 + wave * 16 + r + 8 * hi;   // global token
            int bb = g >> 11, tt = g & (T_ - 1);
            dst[(((size_t)(bb * H_ + hh) * T_ + tt) * D_ + dd)] =
                f2bf(acc[nt][r] * scale);
        }
    }
}

// ---------------------------------------------------------------------------
// Kernel 2: flash attention, one (b,h) head slice, 128 query rows per block.
// ---------------------------------------------------------------------------
__global__ __launch_bounds__(256)
void flash_attn_kernel(const unsigned short* __restrict__ Qg,
                       const unsigned short* __restrict__ Kg,
                       const unsigned short* __restrict__ Vg,
                       unsigned short* __restrict__ Og) {
    __shared__ unsigned short Ks[64][136];     // key tile, row-major, padded
    __shared__ unsigned       VsT[128][36];    // V tile, fragment-major [dim][kpair]
    __shared__ unsigned short Ps[8][16][72];   // per-wave P staging, padded

    const int tid  = threadIdx.x;
    const int lane = tid & 31;
    const int wave = tid >> 5;
    const int m    = lane & 15;
    const int hi   = lane >> 4;

    const int bb = blockIdx.z, hh = blockIdx.y;
    const size_t bh = ((size_t)bb * H_ + hh) * (size_t)T_ * D_;
    const unsigned short* Qb = Qg + bh;
    const unsigned short* Kb = Kg + bh;
    const unsigned short* Vb = Vg + bh;
    const int q0 = blockIdx.x * 128;

    // --- Q fragments, loaded once straight from global (L2-resident) ---
    Frag qf[4];
    {
        const unsigned* qrow =
            (const unsigned*)&Qb[(size_t)(q0 + wave * 16 + m) * D_];
#pragma unroll
        for (int c = 0; c < 4; ++c) {
            int ab = c * 16 + hi * 4;
            *(uint4*)&qf[c].u[0] = *(const uint4*)&qrow[ab];
            *(uint4*)&qf[c].u[4] = *(const uint4*)&qrow[ab + 8];
        }
    }

    v8f oacc[8];
#pragma unroll
    for (int i = 0; i < 8; ++i) oacc[i] = (v8f){};
    float mrow[8], lrow[8];
#pragma unroll
    for (int r = 0; r < 8; ++r) { mrow[r] = -1e30f; lrow[r] = 0.f; }

    auto loadKf = [&](Frag& f, int st, int c) {
        const unsigned* krow = (const unsigned*)&Ks[st * 16 + m][0];
        int kb = c * 16 + hi * 4;
        *(uint4*)&f.u[0] = *(const uint4*)&krow[kb];
        *(uint4*)&f.u[4] = *(const uint4*)&krow[kb + 8];
    };
    auto loadVf = [&](Frag& f, int c, int nt) {
        const unsigned* vrow = &VsT[nt * 16 + m][0];
        int kpb = c * 16 + hi * 4;
        *(uint4*)&f.u[0] = *(const uint4*)&vrow[kpb];
        *(uint4*)&f.u[4] = *(const uint4*)&vrow[kpb + 8];
    };

    for (int s0 = 0; s0 < T_; s0 += 64) {
        __syncthreads();
        for (int i = tid; i < 1024; i += 256) {
            int r = i >> 4, c16 = i & 15;
            ASYNC_CPY16(&Ks[r][c16 * 8], &Kb[(size_t)(s0 + r) * D_ + c16 * 8]);
        }
        for (int i = tid; i < 2048; i += 256) {
            int kp = i >> 6, np = i & 63;
            unsigned a = *(const unsigned*)&Vb[(size_t)(s0 + 2 * kp) * D_ + 2 * np];
            unsigned b = *(const unsigned*)&Vb[(size_t)(s0 + 2 * kp + 1) * D_ + 2 * np];
            VsT[2 * np][kp]     = (a & 0xFFFFu) | (b << 16);
            VsT[2 * np + 1][kp] = (a >> 16) | (b & 0xFFFF0000u);
        }
        ASYNC_WAIT();
        __syncthreads();

        // --- S = Q K^T : double-buffered B stream over (st, c) ---
        v8f sc[4];
#pragma unroll
        for (int st = 0; st < 4; ++st) sc[st] = (v8f){};
        {
            Frag bfr[2];
            loadKf(bfr[0], 0, 0);
#pragma unroll
            for (int i = 0; i < 16; ++i) {
                int st = i >> 2, c = i & 3;
                if (i < 15) loadKf(bfr[(i + 1) & 1], (i + 1) >> 2, (i + 1) & 3);
                sc[st] = WMMA_BF16(qf[c].v, bfr[i & 1].v, sc[st]);
            }
        }

        // --- online softmax (rows live in 16-lane groups) ---
        float anew[8];
#pragma unroll
        for (int r = 0; r < 8; ++r) {
            float mx = fmaxf(fmaxf(sc[0][r], sc[1][r]), fmaxf(sc[2][r], sc[3][r]));
            mx = fmaxf(mx, __shfl_xor(mx, 1, 32));
            mx = fmaxf(mx, __shfl_xor(mx, 2, 32));
            mx = fmaxf(mx, __shfl_xor(mx, 4, 32));
            mx = fmaxf(mx, __shfl_xor(mx, 8, 32));
            float mn = fmaxf(mrow[r], mx);
            float al = __expf(mrow[r] - mn);
            mrow[r] = mn;
            anew[r] = al;
            float ps = 0.f;
#pragma unroll
            for (int st = 0; st < 4; ++st) {
                float p = __expf(sc[st][r] - mn);
                sc[st][r] = p;
                ps += p;
            }
            ps += __shfl_xor(ps, 1, 32);
            ps += __shfl_xor(ps, 2, 32);
            ps += __shfl_xor(ps, 4, 32);
            ps += __shfl_xor(ps, 8, 32);
            lrow[r] = lrow[r] * al + ps;
        }
#pragma unroll
        for (int nt = 0; nt < 8; ++nt)
#pragma unroll
            for (int r = 0; r < 8; ++r) oacc[nt][r] *= anew[r];

        // --- P: C-layout -> A-layout via per-wave LDS staging ---
#pragma unroll
        for (int st = 0; st < 4; ++st)
#pragma unroll
            for (int r = 0; r < 8; ++r)
                Ps[wave][r + 8 * hi][st * 16 + m] = f2bf(sc[st][r]);

        // --- O += P V : hoist P frags, double-buffer V frags ---
        {
            Frag pf[2];
            const unsigned* prow = (const unsigned*)&Ps[wave][m][0];
#pragma unroll
            for (int c = 0; c < 2; ++c) {
                int ab = c * 16 + hi * 4;
                *(uint4*)&pf[c].u[0] = *(const uint4*)&prow[ab];
                *(uint4*)&pf[c].u[4] = *(const uint4*)&prow[ab + 8];
            }
            Frag vfr[2];
            loadVf(vfr[0], 0, 0);
#pragma unroll
            for (int i = 0; i < 16; ++i) {
                int c = i >> 3, nt = i & 7;
                if (i < 15) loadVf(vfr[(i + 1) & 1], (i + 1) >> 3, (i + 1) & 7);
                oacc[nt] = WMMA_BF16(pf[c].v, vfr[i & 1].v, oacc[nt]);
            }
        }
    }

    // epilogue: normalize, write [b,t,h*d] bf16
#pragma unroll
    for (int nt = 0; nt < 8; ++nt)
#pragma unroll
        for (int r = 0; r < 8; ++r) {
            int qr = q0 + wave * 16 + r + 8 * hi;
            Og[((size_t)bb * T_ + qr) * HD_ + hh * D_ + nt * 16 + m] =
                f2bf(oacc[nt][r] / lrow[r]);
        }
}

// ---------------------------------------------------------------------------
// Kernel 3: output projection. out = O @ Wu + bu  (f32 result)
// ---------------------------------------------------------------------------
__global__ __launch_bounds__(256)
void out_proj_kernel(const unsigned short* __restrict__ Ob,
                     const unsigned short* __restrict__ Wub,
                     const float* __restrict__ bu,
                     float* __restrict__ out) {
    __shared__ unsigned short As[128][72];
    __shared__ unsigned       BpT[128][36];

    const int tid  = threadIdx.x;
    const int lane = tid & 31;
    const int wave = tid >> 5;
    const int m    = lane & 15;
    const int hi   = lane >> 4;
    const int m0   = blockIdx.x * 128;

    v8f acc[8];
#pragma unroll
    for (int i = 0; i < 8; ++i) acc[i] = (v8f){};

    auto loadB = [&](Frag& f, int kk, int nt) {
        const unsigned* brow = &BpT[nt * 16 + m][0];
        int kpb = (kk >> 1) + hi * 4;
        *(uint4*)&f.u[0] = *(const uint4*)&brow[kpb];
        *(uint4*)&f.u[4] = *(const uint4*)&brow[kpb + 8];
    };

    for (int kc = 0; kc < HD_; kc += 64) {
        __syncthreads();
        for (int i = tid; i < 1024; i += 256) {
            int r = i >> 3, c4 = i & 7;
            ASYNC_CPY16(&As[r][c4 * 8], &Ob[(size_t)(m0 + r) * HD_ + kc + c4 * 8]);
        }
        for (int i = tid; i < 2048; i += 256) {
            int kp = i >> 6, np = i & 63;
            unsigned a = *(const unsigned*)&Wub[(size_t)(kc + 2 * kp) * D_ + 2 * np];
            unsigned b = *(const unsigned*)&Wub[(size_t)(kc + 2 * kp + 1) * D_ + 2 * np];
            BpT[2 * np][kp]     = (a & 0xFFFFu) | (b << 16);
            BpT[2 * np + 1][kp] = (a >> 16) | (b & 0xFFFF0000u);
        }
        ASYNC_WAIT();
        __syncthreads();

        Frag af[2];
        {
            const unsigned* rowp = (const unsigned*)&As[wave * 16 + m][0];
#pragma unroll
            for (int k2 = 0; k2 < 2; ++k2) {
                int ab = k2 * 16 + hi * 4;
                *(uint4*)&af[k2].u[0] = *(const uint4*)&rowp[ab];
                *(uint4*)&af[k2].u[4] = *(const uint4*)&rowp[ab + 8];
            }
        }
        Frag bfr[2];
        loadB(bfr[0], 0, 0);
#pragma unroll
        for (int i = 0; i < 16; ++i) {
            int nt = i & 7;
            if (i < 15) {
                int j = i + 1;
                loadB(bfr[j & 1], (j >> 3) * 32, j & 7);
            }
            acc[nt] = WMMA_BF16(af[i >> 3].v, bfr[i & 1].v, acc[nt]);
        }
    }

#pragma unroll
    for (int nt = 0; nt < 8; ++nt) {
        int col = nt * 16 + m;
        float bias = bu[col];
#pragma unroll
        for (int r = 0; r < 8; ++r) {
            int g = m0 + wave * 16 + r + 8 * hi;
            out[(size_t)g * D_ + col] = acc[nt][r] + bias;
        }
    }
}

// ---------------------------------------------------------------------------
// Host launcher
// ---------------------------------------------------------------------------
extern "C" void kernel_launch(void* const* d_in, const int* in_sizes, int n_in,
                              void* d_out, int out_size, void* d_ws, size_t ws_size,
                              hipStream_t stream) {
    const float* x  = (const float*)d_in[0];
    const float* Wk = (const float*)d_in[1];
    const float* Wq = (const float*)d_in[2];
    const float* Wv = (const float*)d_in[3];
    const float* Wu = (const float*)d_in[4];
    const float* bu = (const float*)d_in[5];
    float* out = (float*)d_out;
    (void)in_sizes; (void)n_in; (void)out_size; (void)ws_size;

    // workspace layout (bf16 buffers)
    char* w = (char*)d_ws;
    size_t off = 0;
    unsigned short* xb  = (unsigned short*)(w + off); off += (size_t)NTOK * D_ * 2;        // 2 MB
    unsigned short* wkb = (unsigned short*)(w + off); off += (size_t)D_ * HD_ * 2;         // 256 KB
    unsigned short* wqb = (unsigned short*)(w + off); off += (size_t)D_ * HD_ * 2;
    unsigned short* wvb = (unsigned short*)(w + off); off += (size_t)D_ * HD_ * 2;
    unsigned short* wub = (unsigned short*)(w + off); off += (size_t)HD_ * D_ * 2;
    unsigned short* Qb  = (unsigned short*)(w + off); off += (size_t)NTOK * D_ * 2 * H_;   // 16 MB ([b,h,t,d])
    unsigned short* Kb  = (unsigned short*)(w + off); off += (size_t)NTOK * D_ * 2 * H_;
    unsigned short* Vb  = (unsigned short*)(w + off); off += (size_t)NTOK * D_ * 2 * H_;
    unsigned short* Ob  = (unsigned short*)(w + off); off += (size_t)NTOK * HD_ * 2;       // 16 MB

    const float scale = 0.29730177875068026f;   // 128^(-1/4)

    // f32 -> bf16 conversions
    cvt_bf16_kernel<<<1024, 256, 0, stream>>>(x,  xb,  NTOK * D_);
    cvt_bf16_kernel<<<512,  256, 0, stream>>>(Wk, wkb, D_ * HD_);
    cvt_bf16_kernel<<<512,  256, 0, stream>>>(Wq, wqb, D_ * HD_);
    cvt_bf16_kernel<<<512,  256, 0, stream>>>(Wv, wvb, D_ * HD_);
    cvt_bf16_kernel<<<512,  256, 0, stream>>>(Wu, wub, HD_ * D_);

    // QKV projections (reference swaps: queries come from Wk, keys from Wq)
    dim3 gproj(NTOK / 128, HD_ / 128);
    qkv_proj_kernel<<<gproj, 256, 0, stream>>>(xb, wkb, Qb, scale);
    qkv_proj_kernel<<<gproj, 256, 0, stream>>>(xb, wqb, Kb, scale);
    qkv_proj_kernel<<<gproj, 256, 0, stream>>>(xb, wvb, Vb, 1.0f);

    // flash attention
    flash_attn_kernel<<<dim3(T_ / 128, H_, B_), 256, 0, stream>>>(Qb, Kb, Vb, Ob);

    // unify heads
    out_proj_kernel<<<NTOK / 128, 256, 0, stream>>>(Ob, wub, bu, out);
}
